// action_network_84378927497724
// MI455X (gfx1250) — matrix operand order repaired and checked by reading, not compiled, not run on hardware
//
#include <hip/hip_runtime.h>

// ---------------------------------------------------------------------------
// CDNA5 / gfx1250 wave32 WMMA implementation of the hypergraph-GNN forward.
// GEMMs in bf16 (f32 accumulate) via v_wmma_f32_16x16x32_bf16.
// A and B operands are pre-packed into WMMA fragment-major layout so each
// lane's 16-half fragment is one contiguous 32-byte load (2x global_load_b128)
// with zero register repacking.
// ---------------------------------------------------------------------------

typedef __attribute__((ext_vector_type(16))) __bf16 v16bf;
typedef __attribute__((ext_vector_type(8)))  float  v8f;

#define DEV __device__ __forceinline__

#define N_VERTS 50000
#define N_EDGES 10000
#define NNZ     400000
#define DD      256
#define HH      512

// Position of K-index k (0..31) inside a packed 32-half A chunk.
// Low section (bytes 0..31): lanes 0-15, K in {0..7, 16..23}, in that order.
// High section (bytes 32..63): lanes 16-31, K in {8..15, 24..31}.
DEV int pack_pos(int k) {
    return (((k >> 3) & 1) << 4) + (k & 7) + (((k >> 4) & 1) << 3);
}

DEV v8f wmma_bf16(v16bf a, v16bf b, v8f c) {
    return __builtin_amdgcn_wmma_f32_16x16x32_bf16(
        /*neg_a=*/false, a, /*neg_b=*/false, b,
        /*c_mod=*/(short)0, c, /*reuse_a=*/false, /*reuse_b=*/false);
}

// ---- operand packing (f32 -> bf16 fragment-major) --------------------------

// A: [M,K] f32 row-major -> packed [M][K/32][32] bf16 (chunk layout above).
__global__ void pack_a_kernel(const float* __restrict__ src,
                              __bf16* __restrict__ dst, int M, int K) {
    const int total = M * K;
    const int kc = K >> 5;
    for (int i = blockIdx.x * blockDim.x + threadIdx.x; i < total;
         i += gridDim.x * blockDim.x) {
        const int m = i / K, k = i - m * K;
        dst[((long)m * kc + (k >> 5)) * 32 + pack_pos(k & 31)] = (__bf16)src[i];
    }
}

// B: [K,N] f32 row-major -> packed [N/16][K/32][lane(32)][h(16)] bf16.
// lane l holds column n0 + l%16, K = (l<16 ? 0 : 8) + (h & 7) + (h >> 3) * 16.
__global__ void pack_b_kernel(const float* __restrict__ W,
                              __bf16* __restrict__ dst, int K, int N) {
    const int total = K * N;
    const int kc = K >> 5;
    for (int i = blockIdx.x * blockDim.x + threadIdx.x; i < total;
         i += gridDim.x * blockDim.x) {
        const int h    = i & 15;
        const int lane = (i >> 4) & 31;
        const int c    = (i >> 9) % kc;
        const int t    = i / (kc << 9);
        const int k    = (c << 5) + ((lane < 16) ? 0 : 8) + (h & 7) + ((h >> 3) << 4);
        const int n    = (t << 4) + (lane & 15);
        dst[i] = (__bf16)W[(long)k * N + n];
    }
}

// ---- GEMM: out = relu(PA @ PB + bias) --------------------------------------
// block = 256 threads = 8 waves; wave w owns the 16x16 tile at n-tile
// blockIdx.y*8 + w. Inner loop: 4x global_load_b128 + 1x v_wmma.
// OUT_BF16_PACKED: write result in packed-A layout (feeds the next GEMM).

template <bool OUT_BF16_PACKED>
__global__ void gemm_bias_relu_kernel(const __bf16* __restrict__ PA,
                                      const __bf16* __restrict__ PB,
                                      const float* __restrict__ bias,
                                      void* __restrict__ out,
                                      int K, int N) {
    const int lane = threadIdx.x & 31;
    const int wave = threadIdx.x >> 5;
    const int row0 = blockIdx.x * 16;
    const int tile = blockIdx.y * 8 + wave;
    const int kc   = K >> 5;

    const __bf16* ap = PA + ((long)(row0 + (lane & 15)) * kc) * 32
                          + ((lane < 16) ? 0 : 16);
    const __bf16* bp = PB + (((long)tile * kc) * 32 + lane) * 16;

    v8f acc = {};
    for (int c = 0; c < kc; ++c) {
        __builtin_prefetch(ap + (long)(c + 2) * 32, 0, 0);
        v16bf a = *(const v16bf*)(ap + (long)c * 32);
        v16bf b = *(const v16bf*)(bp + (long)c * 512);
        acc = wmma_bf16(a, b, acc);
    }

    const int n    = tile * 16 + (lane & 15);
    const float bn = bias[n];
    const int rtop = (lane < 16) ? 0 : 8;   // C/D layout: VGPR r -> M = r / r+8
#pragma unroll
    for (int r = 0; r < 8; ++r) {
        const int row = row0 + rtop + r;
        float v = fmaxf(acc[r] + bn, 0.0f);
        if (OUT_BF16_PACKED) {
            const int nc = N >> 5;
            ((__bf16*)out)[((long)row * nc + (n >> 5)) * 32 + pack_pos(n & 31)] =
                (__bf16)v;
        } else {
            ((float*)out)[(long)row * N + n] = v;
        }
    }
}

// ---- segment scatter (mean numerators + counts) ----------------------------

__global__ void scatter_v2e_kernel(const float* __restrict__ m2,
                                   const int* __restrict__ vi,
                                   const int* __restrict__ ei,
                                   float* __restrict__ e_acc,
                                   float* __restrict__ e_cnt, int nnz) {
    const int j = threadIdx.x;                 // feature column 0..255
    const int base = blockIdx.x * 8;
#pragma unroll
    for (int t = 0; t < 8; ++t) {
        const int i = base + t;
        if (i >= nnz) return;
        const int v = vi[i], e = ei[i];
        atomicAdd(&e_acc[(long)e * DD + j], m2[(long)v * DD + j]);
        if (j == 0) atomicAdd(&e_cnt[e], 1.0f);
    }
}

// reads edge sums, divides by edge count on the fly (fused e-mean)
__global__ void scatter_e2v_kernel(const float* __restrict__ e_acc,
                                   const float* __restrict__ e_cnt,
                                   const int* __restrict__ vi,
                                   const int* __restrict__ ei,
                                   float* __restrict__ v_acc,
                                   float* __restrict__ v_cnt, int nnz) {
    const int j = threadIdx.x;
    const int base = blockIdx.x * 8;
#pragma unroll
    for (int t = 0; t < 8; ++t) {
        const int i = base + t;
        if (i >= nnz) return;
        const int v = vi[i], e = ei[i];
        const float c = fmaxf(e_cnt[e], 1.0f);
        atomicAdd(&v_acc[(long)v * DD + j], e_acc[(long)e * DD + j] / c);
        if (j == 0) atomicAdd(&v_cnt[v], 1.0f);
    }
}

// ---- fused GEMM3 + residual mean message + ReLU + log_softmax --------------
// block = 256 threads = 8 waves; block owns 16 rows x 256 cols (full rows ->
// log_softmax entirely in LDS, no extra HBM round trip).
// wave w computes n-tiles 2w and 2w+1 (A fragment reused across both WMMAs).

__global__ void gemm3_logsoftmax_kernel(const __bf16* __restrict__ PA,
                                        const __bf16* __restrict__ PB,
                                        const float* __restrict__ bu,
                                        const float* __restrict__ v_acc,
                                        const float* __restrict__ v_cnt,
                                        float* __restrict__ out) {
    __shared__ float tile[16][DD + 8];
    const int lane = threadIdx.x & 31;
    const int wave = threadIdx.x >> 5;
    const int row0 = blockIdx.x * 16;
    const int kc   = DD >> 5;   // 8 chunks

    const __bf16* ap = PA + ((long)(row0 + (lane & 15)) * kc) * 32
                          + ((lane < 16) ? 0 : 16);
    const __bf16* bp0 = PB + (((long)(2 * wave)     * kc) * 32 + lane) * 16;
    const __bf16* bp1 = PB + (((long)(2 * wave + 1) * kc) * 32 + lane) * 16;

    v8f acc0 = {}, acc1 = {};
    for (int c = 0; c < kc; ++c) {
        __builtin_prefetch(ap + (long)(c + 2) * 32, 0, 0);
        v16bf a  = *(const v16bf*)(ap  + (long)c * 32);
        v16bf b0 = *(const v16bf*)(bp0 + (long)c * 512);
        v16bf b1 = *(const v16bf*)(bp1 + (long)c * 512);
        acc0 = wmma_bf16(a, b0, acc0);
        acc1 = wmma_bf16(a, b1, acc1);
    }

    const int rtop = (lane < 16) ? 0 : 8;
#pragma unroll
    for (int t = 0; t < 2; ++t) {
        v8f& acc = t ? acc1 : acc0;
        const int col = wave * 32 + t * 16 + (lane & 15);
        const float bn = bu[col];
#pragma unroll
        for (int r = 0; r < 8; ++r) {
            const int row = row0 + rtop + r;
            const float mi = v_acc[(long)row * DD + col] / fmaxf(v_cnt[row], 1.0f);
            tile[rtop + r][col] = fmaxf(acc[r] + bn + mi, 0.0f);
        }
    }
    __syncthreads();

    // wave w reduces rows 2w and 2w+1; each lane scans 8 columns.
#pragma unroll
    for (int rr = 0; rr < 2; ++rr) {
        const int row = 2 * wave + rr;
        float vals[8];
        float vmax = -3.402823466e38f;
#pragma unroll
        for (int j = 0; j < 8; ++j) {
            vals[j] = tile[row][lane * 8 + j];
            vmax = fmaxf(vmax, vals[j]);
        }
#pragma unroll
        for (int s = 16; s > 0; s >>= 1) vmax = fmaxf(vmax, __shfl_xor(vmax, s, 32));
        float se = 0.0f;
#pragma unroll
        for (int j = 0; j < 8; ++j) se += __expf(vals[j] - vmax);
#pragma unroll
        for (int s = 16; s > 0; s >>= 1) se += __shfl_xor(se, s, 32);
        const float lse = vmax + __logf(se);
        const long base = (long)(row0 + row) * DD + lane * 8;
#pragma unroll
        for (int j = 0; j < 8; ++j) out[base + j] = vals[j] - lse;
    }
}

// ---------------------------------------------------------------------------

static size_t align_up(size_t v) { return (v + 255) & ~(size_t)255; }

extern "C" void kernel_launch(void* const* d_in, const int* in_sizes, int n_in,
                              void* d_out, int out_size, void* d_ws, size_t ws_size,
                              hipStream_t stream) {
    (void)in_sizes; (void)n_in; (void)out_size; (void)ws_size;

    const float* x   = (const float*)d_in[0];
    const int* vi    = (const int*)d_in[1];
    const int* ei    = (const int*)d_in[2];
    const float* W1  = (const float*)d_in[3];
    const float* b1  = (const float*)d_in[4];
    const float* W2  = (const float*)d_in[5];
    const float* b2  = (const float*)d_in[6];
    const float* Wu  = (const float*)d_in[7];
    const float* bu  = (const float*)d_in[8];
    float* out = (float*)d_out;

    // workspace carve-up
    char* p = (char*)d_ws;
    size_t off = 0;
    auto carve = [&](size_t bytes) { void* r = p + off; off = align_up(off + bytes); return r; };
    __bf16* x_pk  = (__bf16*)carve((size_t)N_VERTS * DD * 2);
    __bf16* W1_pk = (__bf16*)carve((size_t)DD * HH * 2);
    __bf16* W2_pk = (__bf16*)carve((size_t)HH * DD * 2);
    __bf16* Wu_pk = (__bf16*)carve((size_t)DD * DD * 2);
    __bf16* m1_pk = (__bf16*)carve((size_t)N_VERTS * HH * 2);
    float*  m2    = (float*) carve((size_t)N_VERTS * DD * 4);
    float*  e_acc = (float*) carve((size_t)N_EDGES * DD * 4);
    float*  e_cnt = (float*) carve((size_t)N_EDGES * 4);
    float*  v_acc = (float*) carve((size_t)N_VERTS * DD * 4);
    float*  v_cnt = (float*) carve((size_t)N_VERTS * 4);

    // zero segment accumulators (per-call: deterministic graph replays)
    hipMemsetAsync(e_acc, 0, (size_t)N_EDGES * DD * 4, stream);
    hipMemsetAsync(e_cnt, 0, (size_t)N_EDGES * 4, stream);
    hipMemsetAsync(v_acc, 0, (size_t)N_VERTS * DD * 4, stream);
    hipMemsetAsync(v_cnt, 0, (size_t)N_VERTS * 4, stream);

    // pack operands into WMMA fragment-major bf16 layouts
    pack_a_kernel<<<2048, 256, 0, stream>>>(x,  x_pk,  N_VERTS, DD);
    pack_b_kernel<<<512,  256, 0, stream>>>(W1, W1_pk, DD, HH);
    pack_b_kernel<<<512,  256, 0, stream>>>(W2, W2_pk, HH, DD);
    pack_b_kernel<<<256,  256, 0, stream>>>(Wu, Wu_pk, DD, DD);

    // m1 = relu(x @ W1 + b1)  [N_VERTS, H], written in packed-A layout
    gemm_bias_relu_kernel<true><<<dim3(N_VERTS / 16, HH / 128), 256, 0, stream>>>(
        x_pk, W1_pk, b1, (void*)m1_pk, DD, HH);

    // m2 = relu(m1 @ W2 + b2) [N_VERTS, D], plain f32 row-major
    gemm_bias_relu_kernel<false><<<dim3(N_VERTS / 16, DD / 128), 256, 0, stream>>>(
        m1_pk, W2_pk, b2, (void*)m2, HH, DD);

    // v -> e scatter (sum + count)
    scatter_v2e_kernel<<<NNZ / 8, DD, 0, stream>>>(m2, vi, ei, e_acc, e_cnt, NNZ);

    // e -> v scatter (divides edge counts on the fly; sum + count)
    scatter_e2v_kernel<<<NNZ / 8, DD, 0, stream>>>(e_acc, e_cnt, vi, ei,
                                                   v_acc, v_cnt, NNZ);

    // h = relu(x @ Wu + bu + v_acc/v_cnt); out = log_softmax(h) fused in LDS
    gemm3_logsoftmax_kernel<<<N_VERTS / 16, 256, 0, stream>>>(
        x_pk, Wu_pk, bu, v_acc, v_cnt, out);
}
